// ginar_cell_16286515986684
// MI455X (gfx1250) — matrix operand order repaired and checked by reading, not compiled, not run on hardware
//
#include <hip/hip_runtime.h>
#include <hip/hip_bf16.h>

// GINAR cell for MI455X (gfx1250, wave32, WMMA).
// Heavy GEMMs use v_wmma_f32_16x16x32_bf16 (bf16 in, f32 accumulate).
// Big N x N operands (L1, L2, P=softmax(relu(ge@geT))) stored bf16 in ws.
// attn = softmax(e) exactly (adj = softmax(..)+I > 0 everywhere, mask no-op).
// graph_learn = I + P folded as  z@graph_learn = z + z@P  (I never formed).
// GEMM: 64x64x64 block tile, 8 waves x (16x32), double-buffered LDS pipeline:
// global->regs for stage k+1 overlaps the 4 WMMAs of stage k; 1 barrier/stage.
// __launch_bounds__(256,2): allow >64 VGPRs so staging regs live across WMMAs.

#define DEVINL __device__ __forceinline__

typedef __attribute__((ext_vector_type(16))) __bf16 v16bf;
typedef __attribute__((ext_vector_type(8)))  float  v8f;

union BfFrag { unsigned short u[16]; v16bf v; };

DEVINL unsigned short f2bf(float f) {
  union { float f; unsigned u; } x; x.f = f;
  unsigned r = x.u + 0x7FFFu + ((x.u >> 16) & 1u);   // round-nearest-even
  return (unsigned short)(r >> 16);
}
DEVINL float bf2f(unsigned short h) {
  union { unsigned u; float f; } x; x.u = ((unsigned)h) << 16; return x.f;
}

// ---------------------------------------------------------------------------
// Generic wave32 WMMA GEMM: C[M,Nc] = A[M,K] * B[K,Nc]  (row-major)
// Block tile 64(M) x 64(N) x 64(K); 8 waves, each a 16x32 C tile
// (2 f32 accumulators, shared A fragment) -> 4 v_wmma per stage.
// Double-buffered LDS software pipeline, one s_barrier per K-stage.
// EPI: 0 = store f32, 1 = +bias[col] store f32, 2 = relu -> bf16 store,
//      3 = accumulate into existing f32 C.
// BBF16: B operand is bf16 in global.  TRANSB: B given as (Nc,K) row-major.
// Requires M%64==0, Nc%64==0, K%64==0.
// ---------------------------------------------------------------------------
template<int EPI, bool BBF16, bool TRANSB>
__global__ __launch_bounds__(256, 2) void gemm_wmma(
    const float* __restrict__ A, const void* __restrict__ Bp, void* __restrict__ Cp,
    const float* __restrict__ bias,
    int M, int Nc, int K, int lda, int ldb, int ldc,
    long long sA, long long sB, long long sC)
{
  (void)M; (void)Nc;
  constexpr int AROW = 72, BROW = 72;          // stride 72 bf16: conflict-free, 8B aligned
  __shared__ unsigned short As[2][64 * AROW];  // ping-pong A tile bf16, [m][k]
  __shared__ unsigned short Bs[2][64 * BROW];  // ping-pong B tile bf16, transposed [n][k]

  const int bz = blockIdx.z;
  A += (long long)bz * sA;
  const float* Bf = nullptr; const unsigned short* Bh = nullptr;
  if constexpr (BBF16) Bh = (const unsigned short*)Bp + (long long)bz * sB;
  else                 Bf = (const float*)Bp + (long long)bz * sB;
  float* Cf = nullptr; unsigned short* Ch = nullptr;
  if constexpr (EPI == 2) Ch = (unsigned short*)Cp + (long long)bz * sC;
  else                    Cf = (float*)Cp + (long long)bz * sC;

  const int m0 = blockIdx.y * 64, n0 = blockIdx.x * 64;
  const int tid = threadIdx.x, lane = tid & 31, wave = tid >> 5;
  const int wm = wave & 3, wn = wave >> 2;     // wave tile: rows wm*16, cols wn*32
  const int half = lane >> 4, l16 = lane & 15;
  const int r = tid >> 2, q16 = (tid & 3) * 16, rot = tid & 3;

  v8f acc[2];
  acc[0] = (v8f){0.f,0.f,0.f,0.f,0.f,0.f,0.f,0.f};
  acc[1] = (v8f){0.f,0.f,0.f,0.f,0.f,0.f,0.f,0.f};

  float aR[16];                 // register staging: A slice (row r, 16 k)
  float bR[16];                 // register staging: B slice (f32 paths)
  unsigned long long bH[4];     // register staging: B slice (bf16 path, 4x8B)

  auto load_stage = [&](int kt) {   // global -> registers (overlaps compute)
    const float* ap = A + (long long)(m0 + r) * lda + kt + q16;
    if (kt + 64 < K) __builtin_prefetch(ap + 64, 0, 1);   // global_prefetch_b8
    #pragma unroll
    for (int i = 0; i < 16; ++i) aR[i] = ap[i];
    if constexpr (TRANSB) {         // B stored (Nc,K): contiguous k per n-row
      const float* bp = Bf + (long long)(n0 + r) * ldb + kt + q16;
      if (kt + 64 < K) __builtin_prefetch(bp + 64, 0, 1);
      #pragma unroll
      for (int i = 0; i < 16; ++i) bR[i] = bp[i];
    } else if constexpr (BBF16) {
      const unsigned short* bp = Bh + (long long)(kt + r) * ldb + n0 + q16;
      if (kt + 64 < K) __builtin_prefetch(bp + 64 * (long long)ldb, 0, 1);
      #pragma unroll
      for (int c = 0; c < 4; ++c) bH[c] = *(const unsigned long long*)(bp + c * 4);
    } else {
      const float* bp = Bf + (long long)(kt + r) * ldb + n0 + q16;
      if (kt + 64 < K) __builtin_prefetch(bp + 64 * (long long)ldb, 0, 1);
      #pragma unroll
      for (int i = 0; i < 16; ++i) bR[i] = bp[i];
    }
  };

  auto store_stage = [&](int buf) { // registers -> LDS[buf] (cvt to bf16)
    unsigned short* Asb = &As[buf][0];
    unsigned short* Bsb = &Bs[buf][0];
    #pragma unroll
    for (int i = 0; i < 16; ++i) Asb[r * AROW + q16 + i] = f2bf(aR[i]);
    if constexpr (TRANSB) {
      #pragma unroll
      for (int i = 0; i < 16; ++i) Bsb[r * BROW + q16 + i] = f2bf(bR[i]);
    } else if constexpr (BBF16) {
      #pragma unroll
      for (int c = 0; c < 4; ++c) {           // rotated order -> conflict-free scatter
        const int ci = (c + rot) & 3;
        const unsigned long long q = bH[ci];
        #pragma unroll
        for (int j = 0; j < 4; ++j)
          Bsb[(q16 + ci * 4 + j) * BROW + r] = (unsigned short)(q >> (16 * j));
      }
    } else {
      #pragma unroll
      for (int c = 0; c < 4; ++c) {
        const int ci = (c + rot) & 3;
        #pragma unroll
        for (int j = 0; j < 4; ++j)
          Bsb[(q16 + ci * 4 + j) * BROW + r] = f2bf(bR[ci * 4 + j]);
      }
    }
  };

  auto compute = [&](int buf) {     // LDS[buf] -> 4 WMMAs
    const unsigned short* Asb = &As[buf][0];
    const unsigned short* Bsb = &Bs[buf][0];
    #pragma unroll
    for (int ks = 0; ks < 2; ++ks) {
      // A frag per ISA: lanes 0-15 row m, K {0..7,16..23}; lanes 16-31 K {8..15,24..31}
      BfFrag af;
      const int ar = (wm * 16 + l16) * AROW + ks * 32;
      #pragma unroll
      for (int i = 0; i < 8; ++i) {
        af.u[i]     = Asb[ar + half * 8 + i];
        af.u[8 + i] = Asb[ar + 16 + half * 8 + i];
      }
      #pragma unroll
      for (int sub = 0; sub < 2; ++sub) {
        // B frag: lane n col, lanes 0-15 K {0..7,16..23}; lanes 16-31 K {8..15,24..31}
        BfFrag bfr;
        const int br = (wn * 32 + sub * 16 + l16) * BROW + ks * 32;
        #pragma unroll
        for (int i = 0; i < 8; ++i) {
          bfr.u[i]     = Bsb[br + half * 8 + i];
          bfr.u[8 + i] = Bsb[br + 16 + half * 8 + i];
        }
        acc[sub] = __builtin_amdgcn_wmma_f32_16x16x32_bf16(false, af.v, false, bfr.v,
                                                           (short)0, acc[sub], false, false);
      }
    }
  };

  // software pipeline: one barrier per K-stage
  load_stage(0);
  store_stage(0);
  __syncthreads();
  int cur = 0;
  for (int kt = 64; kt < K; kt += 64) {
    load_stage(kt);         // in-flight during compute
    compute(cur);
    store_stage(cur ^ 1);   // safe: cur^1 last read 2 stages ago (pre-barrier)
    __syncthreads();
    cur ^= 1;
  }
  compute(cur);

  // C frag: VGPR r -> row r + 8*half; col = l16
  const int row0 = m0 + wm * 16 + half * 8;
  #pragma unroll
  for (int sub = 0; sub < 2; ++sub) {
    const int col = n0 + wn * 32 + sub * 16 + l16;
    float bv = 0.f;
    if constexpr (EPI == 1) bv = bias[col];
    #pragma unroll
    for (int rr = 0; rr < 8; ++rr) {
      float v = acc[sub][rr] + bv;
      long long idx = (long long)(row0 + rr) * ldc + col;
      if constexpr (EPI == 2)      Ch[idx] = f2bf(v > 0.f ? v : 0.f);
      else if constexpr (EPI == 3) Cf[idx] += v;
      else                         Cf[idx] = v;
    }
  }
}

// ---------------------------------------------------------------------------
// Laplacian: d[i] = rsqrt(sum_j(M+I)[i,j]);  L[i,j] = d[i]*(M+I)[j,i]*d[j] (bf16)
// ---------------------------------------------------------------------------
__global__ __launch_bounds__(256) void k_rowsum(const float* __restrict__ graph, float* __restrict__ dv) {
  const int g = blockIdx.y, i = blockIdx.x, t = threadIdx.x;
  const float* row = graph + ((long long)g * 2048 + i) * 2048;
  __shared__ float red[256];
  float s = 0.f;
  for (int j = t; j < 2048; j += 256) s += row[j];
  red[t] = s; __syncthreads();
  for (int o = 128; o > 0; o >>= 1) { if (t < o) red[t] += red[t + o]; __syncthreads(); }
  if (!t) { float d = red[0] + 1.f; dv[g * 2048 + i] = d > 0.f ? rsqrtf(d) : 0.f; }
}

__global__ void k_lap_fill(const float* __restrict__ graph, const float* __restrict__ dv,
                           unsigned short* __restrict__ L1, unsigned short* __restrict__ L2) {
  long long idx = blockIdx.x * 256ll + threadIdx.x;   // 2*N*N
  int j = (int)(idx & 2047);
  int i = (int)((idx >> 11) & 2047);
  int g = (int)(idx >> 22);
  float mv = graph[((long long)g << 22) + (long long)j * 2048 + i] + (i == j ? 1.f : 0.f);
  float v = dv[g * 2048 + i] * mv * dv[g * 2048 + j];
  (g ? L2 : L1)[((long long)i << 11) + j] = f2bf(v);
}

// xe[b,n,e] = leaky_relu(sum_i x[b,i,n]*W[i,e] + b[e])
__global__ void k_xe(const float* __restrict__ x, const float* __restrict__ W,
                     const float* __restrict__ bvec, float* __restrict__ xe) {
  long long idx = blockIdx.x * 256ll + threadIdx.x;   // B*N*E
  int e = (int)(idx & 63);
  long long bn = idx >> 6;
  int n = (int)(bn & 2047);
  long long b = bn >> 11;
  const float* xp = x + b * 12 * 2048 + n;
  float s = bvec[e];
  #pragma unroll
  for (int i = 0; i < 12; ++i) s += xp[(long long)i * 2048] * W[i * 64 + e];
  xe[idx] = s > 0.f ? s : 0.01f * s;
}

// ha1/ha2 = h @ a1 / h @ a2
__global__ void k_ha(const float* __restrict__ h, const float* __restrict__ att_a,
                     float* __restrict__ ha1, float* __restrict__ ha2) {
  int idx = blockIdx.x * 256 + threadIdx.x;           // B*N
  const float* hp = h + (long long)idx * 64;
  float s1 = 0.f, s2 = 0.f;
  #pragma unroll
  for (int e = 0; e < 64; ++e) { float v = hp[e]; s1 += v * att_a[e]; s2 += v * att_a[64 + e]; }
  ha1[idx] = s1; ha2[idx] = s2;
}

__global__ __launch_bounds__(256) void k_ha2max(const float* __restrict__ ha2, float* __restrict__ out) {
  const int b = blockIdx.x, t = threadIdx.x;
  __shared__ float red[256];
  float m = -1e30f;
  for (int n = t; n < 2048; n += 256) m = fmaxf(m, ha2[b * 2048 + n]);
  red[t] = m; __syncthreads();
  for (int o = 128; o > 0; o >>= 1) { if (t < o) red[t] = fmaxf(red[t], red[t + o]); __syncthreads(); }
  if (!t) out[b] = red[0];
}

// Fused attn row: A[b,e,i] = relu( sum_j softmax_j(lrelu(ha1[i]+ha2[j])) * h[b,j,e] )
// Row max is lrelu(ha1[i] + max_j ha2[j]) since lrelu is monotone.
__global__ __launch_bounds__(64) void k_attn(const float* __restrict__ h, const float* __restrict__ ha1,
    const float* __restrict__ ha2, const float* __restrict__ ha2m, float* __restrict__ A) {
  const int b = blockIdx.y, i = blockIdx.x, t = threadIdx.x;
  __shared__ float wb[64];
  const float h1 = ha1[b * 2048 + i];
  float mraw = h1 + ha2m[b];
  const float m = mraw > 0.f ? mraw : 0.01f * mraw;
  float acc = 0.f, sp = 0.f;
  const float* hb = h + (long long)b * 2048 * 64;
  const float* h2 = ha2 + b * 2048;
  for (int j0 = 0; j0 < 2048; j0 += 64) {
    float eraw = h1 + h2[j0 + t];
    float e = eraw > 0.f ? eraw : 0.01f * eraw;
    float w = __expf(e - m);
    wb[t] = w; sp += w;
    __syncthreads();
    const float* hp = hb + (long long)j0 * 64 + t;
    #pragma unroll 8
    for (int jj = 0; jj < 64; ++jj) acc += wb[jj] * hp[jj * 64];
    __syncthreads();
  }
  wb[t] = sp; __syncthreads();
  for (int o = 32; o > 0; o >>= 1) { if (t < o) wb[t] += wb[t + o]; __syncthreads(); }
  float v = acc / wb[0];
  A[((long long)b * 64 + t) * 2048 + i] = v > 0.f ? v : 0.f;   // (B,E,N), relu
}

// geIn[b,n,:] = concat(gl1t[b,:,n], gl2[n,:])
__global__ void k_pack(const float* __restrict__ gl1t, const float* __restrict__ gl2,
                       float* __restrict__ geIn) {
  long long idx = blockIdx.x * 256ll + threadIdx.x;   // B*N*128
  int c = (int)(idx & 127);
  long long bn = idx >> 7;
  int n = (int)(bn & 2047);
  long long b = bn >> 11;
  float v = (c < 64) ? gl1t[(b * 64 + c) * 2048 + n] : gl2[(long long)n * 64 + (c - 64)];
  geIn[idx] = v;
}

// zero-pad cell_GL (2048x32) to (2048x64) so GEMM K%64==0 holds
__global__ void k_padcell(const float* __restrict__ cg, float* __restrict__ out) {
  int idx = blockIdx.x * 256 + threadIdx.x;           // 2048*64
  int k = idx & 63; int m = idx >> 6;
  out[idx] = (k < 32) ? cg[m * 32 + k] : 0.f;
}

// In-place row softmax over bf16 rows of length 2048 (inputs already relu'd >= 0)
__global__ __launch_bounds__(256) void k_softmax_bf16(unsigned short* __restrict__ P) {
  const long long row = blockIdx.x + (long long)blockIdx.y * gridDim.x;
  unsigned short* p = P + row * 2048;
  __shared__ float red[256];
  const int t = threadIdx.x;
  float m = -1e30f;
  for (int n = t; n < 2048; n += 256) m = fmaxf(m, bf2f(p[n]));
  red[t] = m; __syncthreads();
  for (int o = 128; o > 0; o >>= 1) { if (t < o) red[t] = fmaxf(red[t], red[t + o]); __syncthreads(); }
  m = red[0]; __syncthreads();
  float s = 0.f;
  for (int n = t; n < 2048; n += 256) s += __expf(bf2f(p[n]) - m);
  red[t] = s; __syncthreads();
  for (int o = 128; o > 0; o >>= 1) { if (t < o) red[t] += red[t + o]; __syncthreads(); }
  const float inv = 1.f / red[0];
  for (int n = t; n < 2048; n += 256) p[n] = f2bf(__expf(bf2f(p[n]) - m) * inv);
}

// layernorm over last axis of (u + z); optional exact GELU
template<bool GELU>
__global__ __launch_bounds__(256) void k_ln(const float* __restrict__ u, const float* __restrict__ z,
    const float* __restrict__ lnw, const float* __restrict__ lnb, float* __restrict__ out) {
  const long long row = blockIdx.x;                    // B*E rows
  const float* up = u + row * 2048;
  const float* zp = z + row * 2048;
  float* op = out + row * 2048;
  __shared__ float r1[256], r2[256];
  const int t = threadIdx.x;
  float s = 0.f, s2 = 0.f;
  for (int n = t; n < 2048; n += 256) { float v = up[n] + zp[n]; s += v; s2 += v * v; }
  r1[t] = s; r2[t] = s2; __syncthreads();
  for (int o = 128; o > 0; o >>= 1) { if (t < o) { r1[t] += r1[t + o]; r2[t] += r2[t + o]; } __syncthreads(); }
  const float mu  = r1[0] * (1.f / 2048.f);
  const float var = r2[0] * (1.f / 2048.f) - mu * mu;
  const float inv = rsqrtf(var + 1e-5f);
  for (int n = t; n < 2048; n += 256) {
    float v = (up[n] + zp[n] - mu) * inv * lnw[n] + lnb[n];
    if constexpr (GELU) v = 0.5f * v * (1.f + erff(v * 0.70710678118f));
    op[n] = v;
  }
}

// ct_new = g*ct + (1-g)*x_new ; ht = g*elu(ct_new) + (1-g)*xT
__global__ void k_final(const float* __restrict__ gate, const float* __restrict__ xnew,
                        const float* __restrict__ ct, const float* __restrict__ xe,
                        float* __restrict__ out) {
  long long idx = blockIdx.x * 256ll + threadIdx.x;   // B*E*N
  const long long BEN = 8ll * 64 * 2048;
  int n = (int)(idx & 2047);
  long long be = idx >> 11;
  int e = (int)(be & 63);
  long long b = be >> 6;
  float g = gate[idx], xn = xnew[idx];
  float c = g * ct[idx] + xn - g * xn;
  float el = c > 0.f ? c : (__expf(c) - 1.f);
  float xT = xe[(b * 2048 + n) * 64 + e];
  out[idx]       = g * el + xT - g * xT;
  out[BEN + idx] = c;
}

// ---------------------------------------------------------------------------
extern "C" void kernel_launch(void* const* d_in, const int* in_sizes, int n_in,
                              void* d_out, int out_size, void* d_ws, size_t ws_size,
                              hipStream_t stream) {
  (void)in_sizes; (void)n_in; (void)out_size; (void)ws_size;
  const float* x       = (const float*)d_in[0];
  const float* ct      = (const float*)d_in[1];
  const float* graph   = (const float*)d_in[2];
  const float* emb_W   = (const float*)d_in[3];
  const float* emb_b   = (const float*)d_in[4];
  const float* emb2_W  = (const float*)d_in[5];
  const float* emb2_b  = (const float*)d_in[6];
  const float* att_W   = (const float*)d_in[7];
  const float* att_a   = (const float*)d_in[8];
  /* d_in[9] att_GL unused: adj = softmax(..)+I > 0 everywhere -> mask no-op */
  const float* lin1_W  = (const float*)d_in[10];
  const float* lin2_W  = (const float*)d_in[11];
  const float* lin2_b  = (const float*)d_in[12];
  const float* ln_w    = (const float*)d_in[13];
  const float* ln_b    = (const float*)d_in[14];
  const float* cell_GL = (const float*)d_in[15];
  const float* GLlin_W = (const float*)d_in[16];
  const float* GLlin2  = (const float*)d_in[17];

  const long long Nq = 2048, Bq = 8, Eq = 64;
  const long long NN = Nq * Nq, BEN = Bq * Eq * Nq, BNE = Bq * Nq * Eq;

  size_t off = 0;
  unsigned char* base = (unsigned char*)d_ws;
  auto alloc = [&](size_t bytes) -> void* {
    void* p = base + off; off += (bytes + 255) & ~(size_t)255; return p;
  };
  float*          dv    = (float*)alloc(2 * Nq * 4);
  unsigned short* L1b   = (unsigned short*)alloc(NN * 2);
  unsigned short* L2b   = (unsigned short*)alloc(NN * 2);
  float*          xe    = (float*)alloc(BNE * 4);
  float*          hbuf  = (float*)alloc(BNE * 4);
  float*          ha1   = (float*)alloc(Bq * Nq * 4);
  float*          ha2   = (float*)alloc(Bq * Nq * 4);
  float*          ha2m  = (float*)alloc(Bq * 4);
  float*          Abuf  = (float*)alloc(BEN * 4);       // (B,E,N)
  float*          gl1t  = (float*)alloc(BEN * 4);       // (B,E,N)
  float*          gl2   = (float*)alloc(Nq * Eq * 4);
  float*          cellP = (float*)alloc(Nq * 64 * 4);   // cell_GL zero-padded K 32->64
  float*          geIn  = (float*)alloc(Bq * Nq * 128 * 4);
  float*          ge    = (float*)alloc(Bq * Nq * 128 * 4);
  unsigned short* P     = (unsigned short*)alloc(Bq * NN * 2);  // softmax(relu(ge@geT)), bf16
  float*          z1    = (float*)alloc(BEN * 4);
  float*          z2    = (float*)alloc(BEN * 4);
  float*          tA    = (float*)alloc(BEN * 4);
  float*          tB    = (float*)alloc(BEN * 4);
  float*          xnew  = (float*)alloc(BEN * 4);
  float*          gate  = (float*)alloc(BEN * 4);

  // 1. Laplacians -> bf16
  k_rowsum  <<<dim3(2048, 2), 256, 0, stream>>>(graph, dv);
  k_lap_fill<<<(unsigned)(2 * NN / 256), 256, 0, stream>>>(graph, dv, L1b, L2b);

  // 2. xe = lrelu(xT@emb_W+b); h = xe@att_W
  k_xe<<<(unsigned)(BNE / 256), 256, 0, stream>>>(x, emb_W, emb_b, xe);
  gemm_wmma<0,false,false><<<dim3(1, 256, 1), 256, 0, stream>>>(
      xe, att_W, hbuf, nullptr, 16384, 64, 64, 64, 64, 64, 0, 0, 0);

  // 3. attention -> A (B,E,N)
  k_ha    <<<(unsigned)(Bq * Nq / 256), 256, 0, stream>>>(hbuf, att_a, ha1, ha2);
  k_ha2max<<<8, 256, 0, stream>>>(ha2, ha2m);
  k_attn  <<<dim3(2048, 8), 64, 0, stream>>>(hbuf, ha1, ha2, ha2m, Abuf);

  // 4. ge = concat(swap(A@emb2_W+b), cell_GL@GLlin_W) @ GLlin2_W
  gemm_wmma<1,false,false><<<dim3(32, 1, 8), 256, 0, stream>>>(
      Abuf, emb2_W, gl1t, emb2_b, 64, 2048, 2048, 2048, 2048, 2048, Eq * Nq, 0, Eq * Nq);
  k_padcell<<<(unsigned)(Nq * 64 / 256), 256, 0, stream>>>(cell_GL, cellP);
  gemm_wmma<0,false,false><<<dim3(1, 32, 1), 256, 0, stream>>>(
      cellP, GLlin_W, gl2, nullptr, 2048, 64, 64, 64, 64, 64, 0, 0, 0);
  k_pack<<<(unsigned)(Bq * Nq * 128 / 256), 256, 0, stream>>>(gl1t, gl2, geIn);
  gemm_wmma<0,false,false><<<dim3(2, 256, 1), 256, 0, stream>>>(
      geIn, GLlin2, ge, nullptr, 16384, 128, 128, 128, 128, 128, 0, 0, 0);

  // 5. P = softmax(relu(ge@geT)), stored bf16
  gemm_wmma<2,false,true><<<dim3(32, 32, 8), 256, 0, stream>>>(
      ge, ge, P, nullptr, 2048, 2048, 128, 128, 128, 2048, Nq * 128, Nq * 128, NN);
  k_softmax_bf16<<<dim3(2048, 8), 256, 0, stream>>>(P);

  // 6. z1 = A@lin1_W ; z2 = A@lin2_W + lin2_b
  gemm_wmma<0,false,false><<<dim3(32, 1, 8), 256, 0, stream>>>(
      Abuf, lin1_W, z1, nullptr, 64, 2048, 2048, 2048, 2048, 2048, Eq * Nq, 0, Eq * Nq);
  gemm_wmma<1,false,false><<<dim3(32, 1, 8), 256, 0, stream>>>(
      Abuf, lin2_W, z2, lin2_b, 64, 2048, 2048, 2048, 2048, 2048, Eq * Nq, 0, Eq * Nq);

  // 7. branch1: x_new = ln(z1@P + ((z1@L1)@lin1_W)@L2 + z1)
  gemm_wmma<0,true ,false><<<dim3(32, 1, 8), 256, 0, stream>>>(
      z1, L1b, tA, nullptr, 64, 2048, 2048, 2048, 2048, 2048, Eq * Nq, 0, Eq * Nq);
  gemm_wmma<0,false,false><<<dim3(32, 1, 8), 256, 0, stream>>>(
      tA, lin1_W, tB, nullptr, 64, 2048, 2048, 2048, 2048, 2048, Eq * Nq, 0, Eq * Nq);
  gemm_wmma<0,true ,false><<<dim3(32, 1, 8), 256, 0, stream>>>(
      tB, L2b, tA, nullptr, 64, 2048, 2048, 2048, 2048, 2048, Eq * Nq, 0, Eq * Nq);
  gemm_wmma<3,true ,false><<<dim3(32, 1, 8), 256, 0, stream>>>(
      z1, P, tA, nullptr, 64, 2048, 2048, 2048, 2048, 2048, Eq * Nq, NN, Eq * Nq);
  k_ln<false><<<512, 256, 0, stream>>>(tA, z1, ln_w, ln_b, xnew);

  // 8. branch2: gate = gelu(ln(z2@P + ((z2@L1)@lin2_W + lin2_b)@L2 + z2))
  gemm_wmma<0,true ,false><<<dim3(32, 1, 8), 256, 0, stream>>>(
      z2, L1b, tA, nullptr, 64, 2048, 2048, 2048, 2048, 2048, Eq * Nq, 0, Eq * Nq);
  gemm_wmma<1,false,false><<<dim3(32, 1, 8), 256, 0, stream>>>(
      tA, lin2_W, tB, lin2_b, 64, 2048, 2048, 2048, 2048, 2048, Eq * Nq, 0, Eq * Nq);
  gemm_wmma<0,true ,false><<<dim3(32, 1, 8), 256, 0, stream>>>(
      tB, L2b, tA, nullptr, 64, 2048, 2048, 2048, 2048, 2048, Eq * Nq, 0, Eq * Nq);
  gemm_wmma<3,true ,false><<<dim3(32, 1, 8), 256, 0, stream>>>(
      z2, P, tA, nullptr, 64, 2048, 2048, 2048, 2048, 2048, Eq * Nq, NN, Eq * Nq);
  k_ln<true><<<512, 256, 0, stream>>>(tA, z2, ln_w, ln_b, gate);

  // 9. gating -> d_out = [ht, ct_new]
  k_final<<<(unsigned)(BEN / 256), 256, 0, stream>>>(gate, xnew, ct, xe, (float*)d_out);
}
